// PerceiverARAttention_33346126086193
// MI455X (gfx1250) — compile-verified, matrix operand order checked by reading, not run on hardware
//
#include <hip/hip_runtime.h>
#include <hip/hip_bf16.h>

// ---------------- problem constants ----------------
#define BATCH 2
#define SEQ   2048
#define DIM   1024
#define NH    16
#define DH    64
#define LAT   512
#define CTX   1536          // SEQ - LAT
#define BH    (BATCH*NH)    // 32
#define QKV_M (BATCH*SEQ)   // 4096
#define SCALE 0.125f        // DH^-0.5
#define LN_EPS 1e-5f
#define LOG10000 9.210340371976184f

typedef __attribute__((ext_vector_type(16))) _Float16 v16h;
typedef __attribute__((ext_vector_type(8)))  float    v8f;
typedef __attribute__((ext_vector_type(4)))  unsigned u32x4;
typedef __attribute__((ext_vector_type(8)))  unsigned u32x8;

union FragU  { v16h v; float4 f4[2]; };
union Half8  { float4 f4; _Float16 h[8]; };

// D = A*B + C, f16 inputs, f32 accumulate (v_wmma_f32_16x16x32_f16)
__device__ __forceinline__ v8f wmma_f16(v16h a, v16h b, v8f c) {
    return __builtin_amdgcn_wmma_f32_16x16x32_f16(
        /*neg_a=*/false, a, /*neg_b=*/false, b,
        /*c_mod=*/(short)0, c, /*reuse_a=*/false, /*reuse_b=*/false);
}

// Load a 16(rows) x 32(K) f16 fragment from a row-major matrix (K contiguous,
// leading dim `ld` elements). Per CDNA5 ISA 16-bit A layout: lanes 0-15 hold
// K=[0..7] (V0-3) and K=[16..23] (V4-7) of row M=lane; lanes 16-31 hold
// K=[8..15] and K=[24..31] of row M=lane-16. Two b128 loads per lane.
__device__ __forceinline__ v16h load_frag16x32(const _Float16* base, int ld) {
    const int lane = threadIdx.x & 31;
    const int row  = lane & 15;
    const int kk   = (lane >> 4) << 3;    // 0 or 8
    const _Float16* p = base + (size_t)row * ld + kk;
    FragU u;
    u.f4[0] = *(const float4*)(p);        // K = kk .. kk+7
    u.f4[1] = *(const float4*)(p + 16);   // K = kk+16 .. kk+23
    return u.v;
}

// ---------------- TDM: 2D f16 tile -> LDS (Tensor Data Mover) ---------------
// Builds the Tensor DMA Descriptor (D#, CDNA5 ISA) for a
// 2D tile of 16-bit elements and issues tensor_load_to_lds (2 SGPR groups).
// Tracked with TENSORcnt; wave-level DMA (EXEC ignored) -> issue from 1 wave.
__device__ __forceinline__ void tdm_load_tile_f16(
        unsigned lds_off, const void* gaddr,
        unsigned tile_x, unsigned tile_rows,
        unsigned tensor_d0, unsigned tensor_d1, unsigned stride_elems) {
    const unsigned long long ga = (unsigned long long)gaddr;
    u32x4 g0;
    g0[0] = 1u;                                        // count=1, user descriptor
    g0[1] = lds_off;                                   // lds_addr (bytes)
    g0[2] = (unsigned)(ga & 0xffffffffu);              // global_addr[31:0]
    g0[3] = (unsigned)((ga >> 32) & 0x1ffffffu)        // global_addr[56:32]
            | (2u << 30);                              // type = 2 ("image")
    u32x8 g1;
    g1[0] = 1u << 16;                                  // wg_mask=0, data_size=1 (2B)
    g1[1] = (tensor_d0 & 0xffffu) << 16;               // tensor_dim0[15:0]
    g1[2] = (tensor_d0 >> 16) | ((tensor_d1 & 0xffffu) << 16);
    g1[3] = (tensor_d1 >> 16) | (tile_x << 16);        // tile_dim0
    g1[4] = tile_rows;                                 // tile_dim1 (tile_dim2=0)
    g1[5] = stride_elems;                              // tensor_dim0_stride[31:0]
    g1[6] = 0;                                         // stride hi / dim1_stride
    g1[7] = 0;
    asm volatile("tensor_load_to_lds %0, %1" :: "s"(g0), "s"(g1) : "memory");
}

// ---------------- kernel 1: f32 -> f16 convert ----------------
__global__ void cvt_f32_to_f16(const float* __restrict__ in,
                               _Float16* __restrict__ out, int n) {
    int i = blockIdx.x * blockDim.x + threadIdx.x;
    if (i < n) out[i] = (_Float16)in[i];
}

// ---------------- kernel 2: TN WMMA GEMM with TDM-staged A ----------------
// C[M,N] (f32) = A[M,K](f16, row-major) * Bw[N,K](f16, row-major)^T + bias[N]
// Block: 256 threads = 8 waves; block tile 32 rows x 256 cols; each wave owns a
// 16x64 strip. A tile (32 x 64 f16) is DMAed into LDS by the TDM, double
// buffered: wave 0 issues tensor_load_to_lds for chunk i+1, all waves compute
// chunk i from LDS, then s_wait_tensorcnt 0 + barrier hands the buffer over.
__global__ __launch_bounds__(256)
void gemm_tn_f16(const _Float16* __restrict__ A, const _Float16* __restrict__ Bw,
                 float* __restrict__ C, const float* __restrict__ bias,
                 int M, int N, int K) {
    __shared__ _Float16 Atile[2][32][64];   // 8 KB, double buffered

    const int wave  = threadIdx.x >> 5;
    const int lane  = threadIdx.x & 31;
    const int tileN0 = blockIdx.x * 256 + (wave & 3) * 64;
    const int rowSel = (wave >> 2) * 16;                  // 0 or 16 within block tile
    const int tileM  = blockIdx.y * 32 + rowSel;

    const _Float16* Ablock = A + (size_t)blockIdx.y * 32 * K;

    // prologue: DMA first A chunk
    if (wave == 0) {
        tdm_load_tile_f16((unsigned)(unsigned long long)&Atile[0][0][0],
                          Ablock, /*tile_x=*/64, /*rows=*/32,
                          /*tensor_d0=*/K, /*tensor_d1=*/M, /*stride=*/K);
        __builtin_amdgcn_s_wait_tensorcnt(0);
    }
    __syncthreads();

    v8f acc[4] = {};
    for (int kc = 0; kc < K; kc += 64) {
        const int cur = (kc >> 6) & 1;
        if (wave == 0 && kc + 64 < K)
            tdm_load_tile_f16((unsigned)(unsigned long long)&Atile[cur ^ 1][0][0],
                              Ablock + kc + 64, 64, 32, K, M, K);

        const _Float16* Al = &Atile[cur][rowSel][0];
#pragma unroll
        for (int kk = 0; kk < 64; kk += 32) {
            v16h b[4];
#pragma unroll
            for (int t = 0; t < 4; ++t) {
                __builtin_prefetch(Bw + (size_t)(tileN0 + t * 16) * K + kc + 128, 0, 1);
                b[t] = load_frag16x32(Bw + (size_t)(tileN0 + t * 16) * K + kc + kk, K);
            }
            v16h a = load_frag16x32(Al + kk, 64);
#pragma unroll
            for (int t = 0; t < 4; ++t)
                acc[t] = wmma_f16(a, b[t], acc[t]);
        }
        if (wave == 0 && kc + 64 < K)
            __builtin_amdgcn_s_wait_tensorcnt(0);
        __syncthreads();
    }
    // C/D layout: VGPR v, lanes 0-15 -> (M=v, N=lane); lanes 16-31 -> (M=v+8, N=lane-16)
    const int col    = lane & 15;
    const int rowoff = (lane >> 4) * 8;
#pragma unroll
    for (int t = 0; t < 4; ++t) {
        const int n = tileN0 + t * 16 + col;
        const float bv = bias ? bias[n] : 0.0f;
#pragma unroll
        for (int v = 0; v < 8; ++v)
            C[(size_t)(tileM + rowoff + v) * N + n] = acc[t][v] + bv;
    }
}

// ---------------- kernel 3: RoPE (+scale for q) and RoPE+LayerNorm for kv ---
__global__ __launch_bounds__(256)
void rope_ln_kernel(const float* __restrict__ q32, const float* __restrict__ kv32,
                    const float* __restrict__ ln_g, const float* __restrict__ ln_b,
                    _Float16* __restrict__ q16h, _Float16* __restrict__ lkv16h) {
    __shared__ float red[256];
    const int tx  = threadIdx.x;
    const int idx = blockIdx.x * 4 + (tx >> 6);     // enumerates (b, n, h)
    const int d   = tx & 63;
    const int b = idx >> 15;            // / (SEQ*NH)
    const int n = (idx >> 4) & (SEQ-1);
    const int h = idx & (NH-1);

    const size_t src = ((size_t)(b * SEQ + n)) * DIM + h * DH + d;
    const size_t prt = ((size_t)(b * SEQ + n)) * DIM + h * DH + (d ^ 32);

    const float ang = (float)n * __expf(-((float)(d & 31)) * (1.0f / 32.0f) * LOG10000);
    const float c = __cosf(ang), s = __sinf(ang);
    const float sgn = (d < 32) ? -1.0f : 1.0f;

    const float qv  = q32[src]  * c + sgn * q32[prt]  * s;
    const float kvv = kv32[src] * c + sgn * kv32[prt] * s;

    red[tx] = kvv; __syncthreads();
    for (int st = 32; st > 0; st >>= 1) { if ((tx & 63) < st) red[tx] += red[tx + st]; __syncthreads(); }
    const float mean = red[tx & ~63] * (1.0f / 64.0f); __syncthreads();
    const float df = kvv - mean;
    red[tx] = df * df; __syncthreads();
    for (int st = 32; st > 0; st >>= 1) { if ((tx & 63) < st) red[tx] += red[tx + st]; __syncthreads(); }
    const float var = red[tx & ~63] * (1.0f / 64.0f); __syncthreads();

    const float lkv = df * __frsqrt_rn(var + LN_EPS) * ln_g[d] + ln_b[d];

    const size_t dst = ((size_t)(b * NH + h) * SEQ + n) * DH + d;
    q16h[dst]   = (_Float16)(qv * SCALE);
    lkv16h[dst] = (_Float16)lkv;
}

// ---------------- kernel 4: flash attention (1 wave per 16-query tile) ------
__global__ __launch_bounds__(32)
void flash_attn(const _Float16* __restrict__ Q, const _Float16* __restrict__ KV,
                _Float16* __restrict__ Out, int qSeqBase, int mergedBase,
                int nKeys, int causal) {
    __shared__ float    Sbuf[16][32];
    __shared__ _Float16 Pbuf[16][32];
    __shared__ _Float16 Vt[DH][32];
    __shared__ float    rowscale[16];

    const int lane   = threadIdx.x & 31;
    const int col    = lane & 15;
    const int rowoff = (lane >> 4) * 8;
    const int bh = blockIdx.y, b = bh >> 4, h = bh & 15;
    const int q0 = qSeqBase + blockIdx.x * 16;     // absolute query seq of row 0

    const _Float16* qbase = Q + ((size_t)bh * SEQ + q0) * DH;
    const v16h qf0 = load_frag16x32(qbase, DH);
    const v16h qf1 = load_frag16x32(qbase + 32, DH);

    v8f o[4] = {};
    float m_i = -1e30f, l_i = 0.0f;

    for (int kc = 0; kc < nKeys; kc += 32) {
        // stage V^T: lane owns key kc+lane, scatter its 64 dims column-wise
        const _Float16* vrow = KV + ((size_t)bh * SEQ + kc + lane) * DH;
#pragma unroll
        for (int d0 = 0; d0 < DH; d0 += 8) {
            Half8 t; t.f4 = *(const float4*)(vrow + d0);
#pragma unroll
            for (int j = 0; j < 8; ++j) Vt[d0 + j][lane] = t.h[j];
        }
        // S tiles (two 16x16 tiles over this 32-key chunk)
#pragma unroll
        for (int t = 0; t < 2; ++t) {
            const _Float16* kb = KV + ((size_t)bh * SEQ + kc + t * 16) * DH;
            const v16h kb0 = load_frag16x32(kb, DH);
            const v16h kb1 = load_frag16x32(kb + 32, DH);
            v8f s = {};
            s = wmma_f16(qf0, kb0, s);
            s = wmma_f16(qf1, kb1, s);
#pragma unroll
            for (int v = 0; v < 8; ++v) {
                const int m = v + rowoff;
                const int j = kc + t * 16 + col;
                float sv = s[v];
                if (causal && (j > q0 + m)) sv = -1e30f;
                Sbuf[m][t * 16 + col] = sv;
            }
        }
        __syncthreads();
        // online softmax: lanes 0-15 each own one query row
        if (lane < 16) {
            float mx = m_i;
            for (int j = 0; j < 32; ++j) mx = fmaxf(mx, Sbuf[lane][j]);
            const float rs = __expf(m_i - mx);
            float sum = 0.0f;
            for (int j = 0; j < 32; ++j) {
                const float p = __expf(Sbuf[lane][j] - mx);
                sum += p;
                Pbuf[lane][j] = (_Float16)p;
            }
            m_i = mx;
            l_i = l_i * rs + sum;
            rowscale[lane] = rs;
        }
        __syncthreads();
        float rsv[8];
#pragma unroll
        for (int v = 0; v < 8; ++v) rsv[v] = rowscale[v + rowoff];
        const v16h pf = load_frag16x32(&Pbuf[0][0], 32);
#pragma unroll
        for (int t2 = 0; t2 < 4; ++t2) {
#pragma unroll
            for (int v = 0; v < 8; ++v) o[t2][v] *= rsv[v];
            o[t2] = wmma_f16(pf, load_frag16x32(&Vt[t2 * 16][0], 32), o[t2]);
        }
        __syncthreads();
    }
    // normalize and store (merged-head concat layout)
    if (lane < 16) rowscale[lane] = 1.0f / l_i;
    __syncthreads();
    float inv[8];
#pragma unroll
    for (int v = 0; v < 8; ++v) inv[v] = rowscale[v + rowoff];
#pragma unroll
    for (int t2 = 0; t2 < 4; ++t2)
#pragma unroll
        for (int v = 0; v < 8; ++v) {
            const size_t orow = (size_t)b * SEQ + mergedBase + blockIdx.x * 16 + v + rowoff;
            Out[orow * DIM + h * DH + t2 * 16 + col] = (_Float16)(o[t2][v] * inv[v]);
        }
}

// ---------------- launcher ----------------
extern "C" void kernel_launch(void* const* d_in, const int* in_sizes, int n_in,
                              void* d_out, int out_size, void* d_ws, size_t ws_size,
                              hipStream_t stream) {
    const float* x    = (const float*)d_in[0];   // (B,SEQ,DIM)
    const float* Wq   = (const float*)d_in[1];   // (DIM,DIM)
    const float* Wkv  = (const float*)d_in[2];
    const float* Wo   = (const float*)d_in[3];
    const float* bo   = (const float*)d_in[4];   // (DIM,)
    const float* ln_g = (const float*)d_in[5];   // (DH,)
    const float* ln_b = (const float*)d_in[6];

    char* ws = (char*)d_ws;
    _Float16* x16    = (_Float16*)(ws);                       //  8 MB
    _Float16* Wq16   = (_Float16*)(ws + 8388608);             //  2 MB
    _Float16* Wkv16  = (_Float16*)(ws + 10485760);            //  2 MB
    _Float16* Wo16   = (_Float16*)(ws + 12582912);            //  2 MB
    float*    q32    = (float*)   (ws + 14680064);            // 16 MB
    float*    kv32   = (float*)   (ws + 31457280);            // 16 MB
    _Float16* q16h   = (_Float16*)(ws + 48234496);            //  8 MB
    _Float16* lkv16h = (_Float16*)(ws + 56623104);            //  8 MB
    _Float16* out16  = (_Float16*)q32;   // alias: q32 dead after rope_ln

    // 1) convert inputs to f16
    const int NX = QKV_M * DIM, NW = DIM * DIM;
    cvt_f32_to_f16<<<(NX + 255) / 256, 256, 0, stream>>>(x,   x16,  NX);
    cvt_f32_to_f16<<<(NW + 255) / 256, 256, 0, stream>>>(Wq,  Wq16, NW);
    cvt_f32_to_f16<<<(NW + 255) / 256, 256, 0, stream>>>(Wkv, Wkv16, NW);
    cvt_f32_to_f16<<<(NW + 255) / 256, 256, 0, stream>>>(Wo,  Wo16, NW);

    // 2) q = x Wq^T ; kv = x Wkv^T    (4096x1024x1024 each)
    dim3 ggrid(DIM / 256, QKV_M / 32);
    gemm_tn_f16<<<ggrid, 256, 0, stream>>>(x16, Wq16,  q32,  nullptr, QKV_M, DIM, DIM);
    gemm_tn_f16<<<ggrid, 256, 0, stream>>>(x16, Wkv16, kv32, nullptr, QKV_M, DIM, DIM);

    // 3) RoPE + scale (q), RoPE + LayerNorm (kv) -> per-head f16
    rope_ln_kernel<<<(BATCH * SEQ * NH) / 4, 256, 0, stream>>>(q32, kv32, ln_g, ln_b,
                                                               q16h, lkv16h);

    // 4) attention: latent (512 q over 2048 k, causal tail) then ctx (1536 q
    //    over 1536 k, unmasked). Latent output occupies merged rows [0,512).
    flash_attn<<<dim3(LAT / 16, BH), 32, 0, stream>>>(q16h, lkv16h, out16,
                                                      /*qSeqBase=*/CTX, /*mergedBase=*/0,
                                                      /*nKeys=*/SEQ, /*causal=*/1);
    flash_attn<<<dim3(CTX / 16, BH), 32, 0, stream>>>(q16h, lkv16h, out16,
                                                      /*qSeqBase=*/0, /*mergedBase=*/LAT,
                                                      /*nKeys=*/CTX, /*causal=*/0);

    // 5) final projection: d_out = out_all Wo^T + bo  (fp32 output)
    gemm_tn_f16<<<ggrid, 256, 0, stream>>>(out16, Wo16, (float*)d_out, bo,
                                           QKV_M, DIM, DIM);
}
